// FullAttention_69123203662184
// MI455X (gfx1250) — compile-verified, matrix operand order checked by reading, not run on hardware
//
#include <hip/hip_runtime.h>

typedef __attribute__((ext_vector_type(16))) __bf16 v16bf;
typedef __attribute__((ext_vector_type(8)))  __bf16 v8bf;
typedef __attribute__((ext_vector_type(8)))  float  v8f;
typedef __attribute__((ext_vector_type(4)))  unsigned int v4u;
typedef __attribute__((ext_vector_type(8)))  int v8i;
typedef __attribute__((ext_vector_type(4)))  int v4i;

#if defined(__has_builtin)
#if __has_builtin(__builtin_amdgcn_tensor_load_to_lds)
#define HAVE_TDM 1
#endif
#endif
#ifndef HAVE_TDM
#define HAVE_TDM 0
#endif

#define WMMA_BF16(A, B, C) \
  __builtin_amdgcn_wmma_f32_16x16x32_bf16(false, (A), false, (B), (short)0, (C), false, false)

constexpr int Bsz = 4, Lsz = 2048, Hsz = 8, Esz = 64;
constexpr int BLOCK_M = 128;       // 8 waves x 16 rows
constexpr int TILE_S  = 32;        // keys per iteration
constexpr int KSTRIDE = 72;        // 64 bf16 + 8 pad  (144 B rows, 16B aligned)
constexpr int VSTRIDE = 40;        // 32 bf16 + 8 pad  (80 B rows, 16B aligned)
constexpr int ROWSTR  = Hsz * Esz; // l-stride in original [b,l,h,e] layout

__device__ __forceinline__ void splitf(float x, __bf16& hi, __bf16& lo) {
  hi = (__bf16)x;
  lo = (__bf16)(x - (float)hi);
}

__device__ __forceinline__ unsigned ldsoff(const void* p) {
  return (unsigned)(uintptr_t)p;   // low 32 bits of flat LDS pointer = LDS offset
}

__device__ __forceinline__ void async_b128(unsigned lds, const void* gaddr) {
  asm volatile("global_load_async_to_lds_b128 %0, %1, off"
               :: "v"(lds), "v"(gaddr) : "memory");
}

#if HAVE_TDM
// 2D TDM load: TILE_S rows x Esz bf16 elements, row stride Esz, with LDS
// padding 4 DWORDs per 32 DWORDs -> 144-byte LDS rows (KSTRIDE).
__device__ __forceinline__ void tdm_load_ktile(const void* gsrc, unsigned lds) {
  unsigned long long ga = (unsigned long long)(uintptr_t)gsrc;
  v4u g0; v8i g1; v4i gz4 = {0, 0, 0, 0};
  v8i gz8 = {0, 0, 0, 0, 0, 0, 0, 0};
  g0[0] = 1u;                                   // count=1, is_restore=0, no gather
  g0[1] = lds;                                  // lds_addr
  g0[2] = (unsigned)ga;                         // global_addr[31:0]
  g0[3] = (unsigned)((ga >> 32) & 0x01FFFFFFull) | (2u << 30);  // addr[56:32], type=2
  g1[0] = (1 << 16)    // data_size = 2B
        | (1 << 20)    // pad_enable
        | (4 << 22)    // pad_interval = 32 DWORDs (128B)
        | (3 << 25);   // pad_amount   = 4 DWORDs (16B)
  g1[1] = (Esz << 16);           // tensor_dim0[15:0]
  g1[2] = (Lsz << 16);           // tensor_dim1[15:0]
  g1[3] = (Esz << 16);           // tile_dim0
  g1[4] = TILE_S;                // tile_dim1
  g1[5] = Esz;                   // tensor_dim0_stride[31:0]
  g1[6] = 0;
  g1[7] = 0;
  __builtin_amdgcn_tensor_load_to_lds(g0, g1, gz4, gz4, gz8, 0);
}
#endif

// ---------------------------------------------------------------------------
// Pre-pass: split K -> bf16 hi/lo [b,h,l,e]; V -> transposed bf16 hi/lo [b,h,e,l]
// ---------------------------------------------------------------------------
__global__ __launch_bounds__(256)
void preconvert_kernel(const float* __restrict__ k, const float* __restrict__ v,
                       __bf16* __restrict__ khi, __bf16* __restrict__ klo,
                       __bf16* __restrict__ vthi, __bf16* __restrict__ vtlo) {
  const size_t idx8 = ((size_t)blockIdx.x * 256 + threadIdx.x) * 8;
  const int e0 = (int)(idx8 & 63);
  size_t r = idx8 >> 6;                 // (b*L + l)*H + h
  const int h = (int)(r % Hsz);
  const size_t bl = r / Hsz;
  const int l = (int)(bl % Lsz);
  const int b = (int)(bl / Lsz);
  const size_t bh = (size_t)b * Hsz + h;

  const size_t kdst = (bh * Lsz + l) * Esz + e0;
  #pragma unroll
  for (int j = 0; j < 8; ++j) {
    __bf16 hi, lo;
    splitf(k[idx8 + j], hi, lo);
    khi[kdst + j] = hi;
    klo[kdst + j] = lo;
  }
  #pragma unroll
  for (int j = 0; j < 8; ++j) {
    __bf16 hi, lo;
    splitf(v[idx8 + j], hi, lo);
    const size_t vdst = (bh * Esz + (size_t)(e0 + j)) * Lsz + l;
    vthi[vdst] = hi;
    vtlo[vdst] = lo;
  }
}

// ---------------------------------------------------------------------------
// Fast path: pre-converted operands, TDM/async double-buffered LDS staging
// ---------------------------------------------------------------------------
__global__ __launch_bounds__(256)
void attn_fast_kernel(const __bf16* __restrict__ khi, const __bf16* __restrict__ klo,
                      const __bf16* __restrict__ vthi, const __bf16* __restrict__ vtlo,
                      const float* __restrict__ q, float* __restrict__ out) {
  __shared__ __align__(16) __bf16 sKhi[2][TILE_S * KSTRIDE];
  __shared__ __align__(16) __bf16 sKlo[2][TILE_S * KSTRIDE];
  __shared__ __align__(16) __bf16 sVhi[2][Esz * VSTRIDE];
  __shared__ __align__(16) __bf16 sVlo[2][Esz * VSTRIDE];
  __shared__ float sS[8][16 * TILE_S];

  const int tid  = threadIdx.x;
  const int wave = tid >> 5;
  const int lane = tid & 31;
  const int ln   = lane & 15;
  const int g    = lane >> 4;

  const int nmb  = Lsz / BLOCK_M;
  const int mblk = blockIdx.x % nmb;
  const int bh   = blockIdx.x / nmb;
  const int b    = bh / Hsz, h = bh % Hsz;
  const int m0   = mblk * BLOCK_M + wave * 16;

  const __bf16* khbase = khi  + (size_t)bh * Lsz * Esz;
  const __bf16* klbase = klo  + (size_t)bh * Lsz * Esz;
  const __bf16* vhbase = vthi + (size_t)bh * Esz * Lsz;
  const __bf16* vlbase = vtlo + (size_t)bh * Esz * Lsz;

  // ---- Q fragments: 16x64 fp32 -> bf16 hi/lo, A-layout, 2 k-steps
  v16bf qhi[2], qlo[2];
  {
    const float* qrow = q + (((size_t)b * Lsz + (m0 + ln)) * Hsz + h) * Esz;
    #pragma unroll
    for (int ks = 0; ks < 2; ++ks) {
      #pragma unroll
      for (int i = 0; i < 16; ++i) {
        const int e = ks * 32 + ((i >> 3) << 4) + (i & 7) + 8 * g;
        __bf16 hi, lo;
        splitf(qrow[e], hi, lo);
        qhi[ks][i] = hi;
        qlo[ks][i] = lo;
      }
    }
  }

  v8f acc[4] = {};

  // per-thread async mapping: 16 bytes per instruction
  const int vrow = tid >> 2;        // e-row 0..63 (64B payload per row)
  const int vchk = (tid & 3) * 16;  // byte chunk within row
#if !HAVE_TDM
  const int krow = tid >> 3;        // key row 0..31 (128B payload per row)
  const int kchk = (tid & 7) * 16;
#endif

  auto issue_tile = [&](int buf, int s) {
#if HAVE_TDM
    if (wave == 0) {
      tdm_load_ktile(khbase + (size_t)s * Esz, ldsoff(&sKhi[buf][0]));
      tdm_load_ktile(klbase + (size_t)s * Esz, ldsoff(&sKlo[buf][0]));
    }
#else
    async_b128(ldsoff(&sKhi[buf][0]) + krow * (KSTRIDE * 2) + kchk,
               (const char*)(khbase + (size_t)(s + krow) * Esz) + kchk);
    async_b128(ldsoff(&sKlo[buf][0]) + krow * (KSTRIDE * 2) + kchk,
               (const char*)(klbase + (size_t)(s + krow) * Esz) + kchk);
#endif
    async_b128(ldsoff(&sVhi[buf][0]) + vrow * (VSTRIDE * 2) + vchk,
               (const char*)(vhbase + (size_t)vrow * Lsz + s) + vchk);
    async_b128(ldsoff(&sVlo[buf][0]) + vrow * (VSTRIDE * 2) + vchk,
               (const char*)(vlbase + (size_t)vrow * Lsz + s) + vchk);
  };

  auto wait_tile = [&]() {
#if HAVE_TDM
    __builtin_amdgcn_s_wait_tensorcnt(0);
#endif
    asm volatile("s_wait_asynccnt 0x0" ::: "memory");
  };

  // prologue: fill buffer 0
  issue_tile(0, 0);
  wait_tile();
  __syncthreads();

  int p = 0;
  for (int s = 0; s < Lsz; s += TILE_S) {
    // start DMA of next tile into the other buffer; overlaps compute below
    if (s + TILE_S < Lsz) issue_tile(p ^ 1, s + TILE_S);

    // ---- GEMM1: S[16x32] = Q * K_tile^T (bf16x3 split)
    #pragma unroll
    for (int t = 0; t < 2; ++t) {
      v8f c = {};
      #pragma unroll
      for (int ks = 0; ks < 2; ++ks) {
        const int e0 = ks * 32;
        const v8bf* ph = (const v8bf*)&sKhi[p][(t * 16 + ln) * KSTRIDE + e0 + 8 * g];
        const v8bf* pl = (const v8bf*)&sKlo[p][(t * 16 + ln) * KSTRIDE + e0 + 8 * g];
        const v8bf h0 = ph[0], h1 = ph[2];
        const v8bf l0 = pl[0], l1 = pl[2];
        v16bf bhv, blv;
        #pragma unroll
        for (int i = 0; i < 8; ++i) {
          bhv[i] = h0[i]; bhv[8 + i] = h1[i];
          blv[i] = l0[i]; blv[8 + i] = l1[i];
        }
        c = WMMA_BF16(qhi[ks], bhv, c);
        c = WMMA_BF16(qhi[ks], blv, c);
        c = WMMA_BF16(qlo[ks], bhv, c);
      }
      #pragma unroll
      for (int r = 0; r < 8; ++r)
        sS[wave][(r + 8 * g) * TILE_S + t * 16 + ln] = c[r];
    }

    // ---- S: C-layout -> A-layout via per-wave LDS staging, re-split
    v16bf ahi, alo;
    #pragma unroll
    for (int i = 0; i < 16; ++i) {
      const int kk = ((i >> 3) << 4) + (i & 7) + 8 * g;
      __bf16 hi, lo;
      splitf(sS[wave][ln * TILE_S + kk], hi, lo);
      ahi[i] = hi;
      alo[i] = lo;
    }

    // ---- GEMM2: acc += S * V_tile (bf16x3 split)
    #pragma unroll
    for (int t = 0; t < 4; ++t) {
      const v8bf* ph = (const v8bf*)&sVhi[p][(t * 16 + ln) * VSTRIDE + 8 * g];
      const v8bf* pl = (const v8bf*)&sVlo[p][(t * 16 + ln) * VSTRIDE + 8 * g];
      const v8bf h0 = ph[0], h1 = ph[2];
      const v8bf l0 = pl[0], l1 = pl[2];
      v16bf bhv, blv;
      #pragma unroll
      for (int i = 0; i < 8; ++i) {
        bhv[i] = h0[i]; bhv[8 + i] = h1[i];
        blv[i] = l0[i]; blv[8 + i] = l1[i];
      }
      acc[t] = WMMA_BF16(ahi, bhv, acc[t]);
      acc[t] = WMMA_BF16(ahi, blv, acc[t]);
      acc[t] = WMMA_BF16(alo, bhv, acc[t]);
    }

    wait_tile();        // next buffer's DMA finished
    __syncthreads();    // publish to all waves / protect reuse
    p ^= 1;
  }

  // ---- epilogue
  float* obase = out + (((size_t)b * Hsz + h) * Lsz + m0) * Esz;
  #pragma unroll
  for (int t = 0; t < 4; ++t) {
    #pragma unroll
    for (int r = 0; r < 8; ++r)
      obase[(size_t)(r + 8 * g) * Esz + t * 16 + ln] = acc[t][r];
  }
}

// ---------------------------------------------------------------------------
// Fallback (no workspace): convert-in-kernel staging (round-1 kernel)
// ---------------------------------------------------------------------------
__global__ __launch_bounds__(256)
void attn_fallback_kernel(const float* __restrict__ q, const float* __restrict__ k,
                          const float* __restrict__ v, float* __restrict__ out) {
  __shared__ __align__(16) __bf16 sKhi[TILE_S * KSTRIDE];
  __shared__ __align__(16) __bf16 sKlo[TILE_S * KSTRIDE];
  __shared__ __align__(16) __bf16 sVhi[Esz * VSTRIDE];
  __shared__ __align__(16) __bf16 sVlo[Esz * VSTRIDE];
  __shared__ float sS[8][16 * TILE_S];

  const int tid  = threadIdx.x;
  const int wave = tid >> 5;
  const int lane = tid & 31;
  const int ln   = lane & 15;
  const int g    = lane >> 4;

  const int nmb  = Lsz / BLOCK_M;
  const int mblk = blockIdx.x % nmb;
  const int bh   = blockIdx.x / nmb;
  const int b    = bh / Hsz, h = bh % Hsz;
  const int m0   = mblk * BLOCK_M + wave * 16;

  v16bf qhi[2], qlo[2];
  {
    const float* qrow = q + (((size_t)b * Lsz + (m0 + ln)) * Hsz + h) * Esz;
    #pragma unroll
    for (int ks = 0; ks < 2; ++ks) {
      #pragma unroll
      for (int i = 0; i < 16; ++i) {
        const int e = ks * 32 + ((i >> 3) << 4) + (i & 7) + 8 * g;
        __bf16 hi, lo;
        splitf(qrow[e], hi, lo);
        qhi[ks][i] = hi;
        qlo[ks][i] = lo;
      }
    }
  }

  v8f acc[4] = {};
  const float* kbase = k + (((size_t)b * Lsz) * Hsz + h) * Esz;
  const float* vbase = v + (((size_t)b * Lsz) * Hsz + h) * Esz;

  const int st_idx = tid * 8;
  const int st_kr  = st_idx >> 6;
  const int st_e   = st_idx & 63;

  for (int s = 0; s < Lsz; s += TILE_S) {
    __syncthreads();
    {
      const float* srck = kbase + (size_t)(s + st_kr) * ROWSTR + st_e;
      const float* srcv = vbase + (size_t)(s + st_kr) * ROWSTR + st_e;
      #pragma unroll
      for (int j = 0; j < 8; ++j) {
        __bf16 hi, lo;
        splitf(srck[j], hi, lo);
        sKhi[st_kr * KSTRIDE + st_e + j] = hi;
        sKlo[st_kr * KSTRIDE + st_e + j] = lo;
      }
      #pragma unroll
      for (int j = 0; j < 8; ++j) {
        __bf16 hi, lo;
        splitf(srcv[j], hi, lo);
        sVhi[(st_e + j) * VSTRIDE + st_kr] = hi;
        sVlo[(st_e + j) * VSTRIDE + st_kr] = lo;
      }
    }
    if (s + TILE_S < Lsz) {
      __builtin_prefetch(kbase + (size_t)(s + TILE_S + st_kr) * ROWSTR + st_e, 0, 3);
      __builtin_prefetch(vbase + (size_t)(s + TILE_S + st_kr) * ROWSTR + st_e, 0, 3);
    }
    __syncthreads();

    #pragma unroll
    for (int t = 0; t < 2; ++t) {
      v8f c = {};
      #pragma unroll
      for (int ks = 0; ks < 2; ++ks) {
        const int e0 = ks * 32;
        const v8bf* ph = (const v8bf*)&sKhi[(t * 16 + ln) * KSTRIDE + e0 + 8 * g];
        const v8bf* pl = (const v8bf*)&sKlo[(t * 16 + ln) * KSTRIDE + e0 + 8 * g];
        const v8bf h0 = ph[0], h1 = ph[2];
        const v8bf l0 = pl[0], l1 = pl[2];
        v16bf bhv, blv;
        #pragma unroll
        for (int i = 0; i < 8; ++i) {
          bhv[i] = h0[i]; bhv[8 + i] = h1[i];
          blv[i] = l0[i]; blv[8 + i] = l1[i];
        }
        c = WMMA_BF16(qhi[ks], bhv, c);
        c = WMMA_BF16(qhi[ks], blv, c);
        c = WMMA_BF16(qlo[ks], bhv, c);
      }
      #pragma unroll
      for (int r = 0; r < 8; ++r)
        sS[wave][(r + 8 * g) * TILE_S + t * 16 + ln] = c[r];
    }

    v16bf ahi, alo;
    #pragma unroll
    for (int i = 0; i < 16; ++i) {
      const int kk = ((i >> 3) << 4) + (i & 7) + 8 * g;
      __bf16 hi, lo;
      splitf(sS[wave][ln * TILE_S + kk], hi, lo);
      ahi[i] = hi;
      alo[i] = lo;
    }

    #pragma unroll
    for (int t = 0; t < 4; ++t) {
      const v8bf* ph = (const v8bf*)&sVhi[(t * 16 + ln) * VSTRIDE + 8 * g];
      const v8bf* pl = (const v8bf*)&sVlo[(t * 16 + ln) * VSTRIDE + 8 * g];
      const v8bf h0 = ph[0], h1 = ph[2];
      const v8bf l0 = pl[0], l1 = pl[2];
      v16bf bhv, blv;
      #pragma unroll
      for (int i = 0; i < 8; ++i) {
        bhv[i] = h0[i]; bhv[8 + i] = h1[i];
        blv[i] = l0[i]; blv[8 + i] = l1[i];
      }
      acc[t] = WMMA_BF16(ahi, bhv, acc[t]);
      acc[t] = WMMA_BF16(ahi, blv, acc[t]);
      acc[t] = WMMA_BF16(alo, bhv, acc[t]);
    }
  }

  float* obase = out + (((size_t)b * Hsz + h) * Lsz + m0) * Esz;
  #pragma unroll
  for (int t = 0; t < 4; ++t) {
    #pragma unroll
    for (int r = 0; r < 8; ++r)
      obase[(size_t)(r + 8 * g) * Esz + t * 16 + ln] = acc[t][r];
  }
}

extern "C" void kernel_launch(void* const* d_in, const int* in_sizes, int n_in,
                              void* d_out, int out_size, void* d_ws, size_t ws_size,
                              hipStream_t stream) {
  const float* q = (const float*)d_in[0];
  const float* k = (const float*)d_in[1];
  const float* v = (const float*)d_in[2];
  float* out = (float*)d_out;
  (void)in_sizes; (void)n_in; (void)out_size;

  const int grid = Bsz * Hsz * (Lsz / BLOCK_M);  // 512 workgroups
  constexpr size_t NELEM = (size_t)Bsz * Hsz * Lsz * Esz;   // 4M per tensor
  constexpr size_t WS_NEED = NELEM * 2 * 4;                 // 4 bf16 arrays = 32MB

  if (d_ws != nullptr && ws_size >= WS_NEED) {
    __bf16* khi  = (__bf16*)d_ws;
    __bf16* klo  = khi + NELEM;
    __bf16* vthi = klo + NELEM;
    __bf16* vtlo = vthi + NELEM;
    const int cgrid = (int)(NELEM / 8 / 256);   // 2048 blocks
    preconvert_kernel<<<cgrid, 256, 0, stream>>>(k, v, khi, klo, vthi, vtlo);
    attn_fast_kernel<<<grid, 256, 0, stream>>>(khi, klo, vthi, vtlo, q, out);
  } else {
    attn_fallback_kernel<<<grid, 256, 0, stream>>>(q, k, v, out);
  }
}
